// Voxel2Mesh_32220844655015
// MI455X (gfx1250) — matrix operand order repaired and checked by reading, not compiled, not run on hardware
//
#include <hip/hip_runtime.h>
#include <hip/hip_bf16.h>

// ===========================================================================
// Voxel2Mesh forward for MI455X (gfx1250, wave32, WMMA).
//
// Roofline: ~24 GFLOP of 3x3x3 convs over ~60MB of activations -> entire
// working set is L2-resident (192MB L2); feed the matrix pipes with
// v_wmma_f32_16x16x32_f16 (f16 in, f32 accumulate).  Activations are stored
// f16 channel-last so the GEMM K-dim (channels) is contiguous (16B fragment
// loads) and channel-concat is free (write into channel sub-ranges).
//
// Input flattening assumption (jax tree flatten = dict keys sorted at every
// level): top level -> cnn, edges0, faces0, faces1, gnn, verts0, x.
//   cnn: enc[0..2]{Wa,Wb,ba,bb}, final{W,b}, up[0..1]{Wa,Wb,Wt,ba,bb,bt}
//   gnn: f2f[3][3]{W0,W1,b}, f2v[3][3]{W0,W1,b}
// ===========================================================================

typedef __attribute__((ext_vector_type(16))) _Float16 v16h;
typedef __attribute__((ext_vector_type(8)))  _Float16 h8;
typedef __attribute__((ext_vector_type(8)))  float    v8f;

static inline int cdiv(int a, int b) { return (a + b - 1) / b; }

// ---------------- input indices (see flattening assumption above) ----------
enum {
  IDX_E0_WA = 0, IDX_E0_WB = 1, IDX_E0_BA = 2, IDX_E0_BB = 3,
  IDX_E1_WA = 4, IDX_E1_WB = 5, IDX_E1_BA = 6, IDX_E1_BB = 7,
  IDX_E2_WA = 8, IDX_E2_WB = 9, IDX_E2_BA = 10, IDX_E2_BB = 11,
  IDX_FIN_W = 12, IDX_FIN_B = 13,
  IDX_U0_WA = 14, IDX_U0_WB = 15, IDX_U0_WT = 16, IDX_U0_BA = 17, IDX_U0_BB = 18, IDX_U0_BT = 19,
  IDX_U1_WA = 20, IDX_U1_WB = 21, IDX_U1_WT = 22, IDX_U1_BA = 23, IDX_U1_BB = 24, IDX_U1_BT = 25,
  IDX_EDGES = 26, IDX_FACES0 = 27, IDX_FACES1 = 28,
  IDX_F2F = 29,   // 27 tensors: f2f[s][l]{W0,W1,b}
  IDX_F2V = 56,   // 27 tensors: f2v[s][l]{W0,W1,b}
  IDX_VERTS0 = 83, IDX_X = 84
};

// ---------------------------------------------------------------------------
// f32 -> f16 conversion
// ---------------------------------------------------------------------------
__global__ void f32_to_f16_k(const float* __restrict__ s, _Float16* __restrict__ d, int n) {
  int i = blockIdx.x * blockDim.x + threadIdx.x;
  if (i < n) d[i] = (_Float16)s[i];
}

// ---------------------------------------------------------------------------
// Pack conv weights (Cout,Cin,3,3,3) f32 into per-lane WMMA B-fragments:
// layout [tap][kchunk][ntile][lane][16 f16]  (one aligned 32B load per frag).
// B 32x16 f16 order used here (contiguous k, as A): lane n (0..15) holds K=0..15 of column n (2 f16/VGPR),
// lane n+16 holds K=16..31 of column n.  K padded to 32 with zeros.
// ---------------------------------------------------------------------------
__global__ void pack_w_k(const float* __restrict__ W, _Float16* __restrict__ out,
                         int Cin, int Cout) {
  int KC = (Cin + 31) / 32, NT = Cout / 16;
  int total = 27 * KC * NT * 32;
  int idx = blockIdx.x * blockDim.x + threadIdx.x;
  if (idx >= total) return;
  int lane = idx & 31;
  int r = idx >> 5;
  int nt = r % NT; r /= NT;
  int kc = r % KC; int t = r / KC;
  int n = nt * 16 + (lane & 15);
  int kbase = kc * 32 + ((lane < 16) ? 0 : 16);
  _Float16* o = out + (size_t)idx * 16;
  for (int e = 0; e < 16; ++e) {
    int k = kbase + e;
    float v = (k < Cin) ? W[((size_t)n * Cin + k) * 27 + t] : 0.0f;
    o[e] = (_Float16)v;
  }
}

// ---------------------------------------------------------------------------
// 3x3x3 SAME conv as tap-decomposed implicit GEMM on WMMA.
// One wave = one 16(spatial) x 16(Cout) output tile; K = channels (padded 32).
// A 16x32 f16 order used here (contiguous k): lanes 0..15 -> M=lane, K 0..15;
//                     lanes 16..31 -> M=lane-16, K 16..31.
// Boundary handling: clamp the voxel address (always in-bounds load), then
// zero the fragment via select -- no divergent branches in the hot loop.
// Tap loop is three fully-unrolled dz/dy/dx loops so tap indices and wp
// offsets are compile-time constants and clamps hoist per-axis.  INS is a
// template constant (power of two) and all indices are 32-bit so address
// math is shifts + saddr-form loads, not 64-bit multiplies.
// ---------------------------------------------------------------------------
template <int S, int CIN, int COUT, int INS, bool RELU>
__global__ __launch_bounds__(256) void conv3d_wmma_k(
    const _Float16* __restrict__ act,
    const _Float16* __restrict__ wp,
    const float* __restrict__ bias,
    _Float16* __restrict__ out, int outStride, int outOfs) {
  constexpr int M  = S * S * S;
  constexpr int MT = M / 16;
  constexpr int NT = COUT / 16;
  constexpr int KC = (CIN + 31) / 32;
  int wave = blockIdx.x * (blockDim.x >> 5) + (threadIdx.x >> 5);
  if (wave >= MT * NT) return;            // wave-uniform: EXEC all-1 at WMMA
  int lane = threadIdx.x & 31;
  int nt = wave % NT;
  int mt = wave / NT;
  int m = lane & 15;
  int pos = mt * 16 + m;
  int px = pos % S, py = (pos / S) % S, pz = pos / (S * S);
  bool hiHalf = (lane >= 16);
  int kgrp = hiHalf ? 16 : 0;
  int ncol = lane & 15;
  float bv = bias[nt * 16 + ncol];
  v8f acc = {bv, bv, bv, bv, bv, bv, bv, bv};
  const h8 zeroh = {};

#pragma unroll
  for (int dz = -1; dz <= 1; ++dz) {
    int zz = pz + dz;
    bool okz = (unsigned)zz < (unsigned)S;
    int zc = zz < 0 ? 0 : (zz > S - 1 ? S - 1 : zz);
#pragma unroll
    for (int dy = -1; dy <= 1; ++dy) {
      int yy = py + dy;
      bool oky = okz & ((unsigned)yy < (unsigned)S);
      int yc = yy < 0 ? 0 : (yy > S - 1 ? S - 1 : yy);
      int rowIdx = (zc * S + yc) * S * INS;            // 32-bit, shifts only
#pragma unroll
      for (int dx = -1; dx <= 1; ++dx) {
        int xx = px + dx;
        bool ok = oky & ((unsigned)xx < (unsigned)S);
        int xc = xx < 0 ? 0 : (xx > S - 1 ? S - 1 : xx);
        const _Float16* src = act + (rowIdx + xc * INS);
        const int t = (dz + 1) * 9 + (dy + 1) * 3 + (dx + 1);   // constant
#pragma unroll
        for (int kc = 0; kc < KC; ++kc) {
          v16h a;
          if constexpr (CIN == 1) {
            _Float16 v0 = src[0];                    // unconditional, clamped
            v16h az = {};
            az[0] = (ok && !hiHalf) ? v0 : (_Float16)0.0f;
            a = az;
          } else {
            // group 0: K = kc*32 + kgrp + 0..7  (zero pad when CIN < chunk)
            h8 g0 = zeroh;
            if (CIN >= kc * 32 + kgrp + 8) {
              h8 t0 = *(const h8*)(src + kc * 32 + kgrp);   // 16B vector load
              g0 = ok ? t0 : zeroh;
            }
            // group 1: K = kc*32 + kgrp + 8..15 (zero pad when CIN < chunk)
            h8 g1 = zeroh;
            if (CIN >= kc * 32 + kgrp + 16) {
              h8 t1 = *(const h8*)(src + kc * 32 + kgrp + 8);
              g1 = ok ? t1 : zeroh;
            }
#pragma unroll
            for (int e = 0; e < 8; ++e) { a[e] = g0[e]; a[e + 8] = g1[e]; }
          }
          v16h bf = *(const v16h*)(wp + (((t * KC + kc) * NT + nt) * 32 + lane) * 16);
          acc = __builtin_amdgcn_wmma_f32_16x16x32_f16(
              /*neg_a=*/false, a, /*neg_b=*/false, bf,
              /*c_mod=*/(short)0, acc, /*reuse_a=*/false, /*reuse_b=*/false);
        }
      }
    }
  }

  int rbase = (lane < 16) ? 0 : 8;
#pragma unroll
  for (int r = 0; r < 8; ++r) {
    float v = acc[r];
    if (RELU) v = v > 0.0f ? v : 0.0f;
    int opos = mt * 16 + rbase + r;
    out[opos * outStride + outOfs + nt * 16 + ncol] = (_Float16)v;
  }
}

// ---------------------------------------------------------------------------
// 2x2x2 maxpool, channel-last
// ---------------------------------------------------------------------------
__global__ void maxpool_k(const _Float16* __restrict__ in, int inStride, int C, int S,
                          _Float16* __restrict__ out) {
  int OS = S / 2;
  int total = OS * OS * OS * C;
  int idx = blockIdx.x * blockDim.x + threadIdx.x;
  if (idx >= total) return;
  int c = idx % C; int p = idx / C;
  int ox = p % OS, oy = (p / OS) % OS, oz = p / (OS * OS);
  float mx = -3.4e38f;
  for (int a = 0; a < 2; ++a)
    for (int b = 0; b < 2; ++b)
      for (int d = 0; d < 2; ++d) {
        int iz = 2 * oz + a, iy = 2 * oy + b, ix = 2 * ox + d;
        float v = (float)in[(size_t)((iz * S + iy) * S + ix) * inStride + c];
        mx = v > mx ? v : mx;
      }
  out[(size_t)p * C + c] = (_Float16)mx;
}

// ---------------------------------------------------------------------------
// Transposed conv 2x2x2 stride 2 (einsum 'ncdhw,coijk->nodihjwk').
// Each output voxel maps to exactly one input voxel + one tap.
// ---------------------------------------------------------------------------
__global__ void upconv_k(const _Float16* __restrict__ in, int CIN, int S,
                         const float* __restrict__ Wt, const float* __restrict__ bt,
                         int COUT, _Float16* __restrict__ out, int outStride, int outOfs) {
  int OS = 2 * S;
  int total = OS * OS * OS * COUT;
  int idx = blockIdx.x * blockDim.x + threadIdx.x;
  if (idx >= total) return;
  int o = idx % COUT; int p = idx / COUT;
  int ox = p % OS, oy = (p / OS) % OS, oz = p / (OS * OS);
  int tap = ((oz & 1) << 2) | ((oy & 1) << 1) | (ox & 1);
  const _Float16* src = in + (size_t)(((oz >> 1) * S + (oy >> 1)) * S + (ox >> 1)) * CIN;
  float acc = bt[o];
  for (int c = 0; c < CIN; ++c)
    acc += (float)src[c] * Wt[((size_t)c * COUT + o) * 8 + tap];
  out[(size_t)p * outStride + outOfs + o] = (_Float16)acc;
}

// ---------------------------------------------------------------------------
// Final 1x1x1 conv 16 -> 2, f32 output in NCDHW order
// ---------------------------------------------------------------------------
__global__ void final_conv_k(const _Float16* __restrict__ h, const float* __restrict__ W,
                             const float* __restrict__ b, float* __restrict__ out, int N) {
  int idx = blockIdx.x * blockDim.x + threadIdx.x;
  if (idx >= 2 * N) return;
  int o = idx / N; int p = idx % N;
  const _Float16* src = h + (size_t)p * 16;
  float acc = b[o];
  for (int c = 0; c < 16; ++c) acc += (float)src[c] * W[o * 16 + c];
  out[(size_t)o * N + p] = acc;
}

// ---------------------------------------------------------------------------
// Trilinear sampling of a channel-last cube (S^3, stride CS, C channels).
// ---------------------------------------------------------------------------
__global__ void trilinear_k(const _Float16* __restrict__ feat, int S, int CS, int C,
                            const float* __restrict__ verts, int NV,
                            float* __restrict__ out, int outStride, int outOfs) {
  int idx = blockIdx.x * blockDim.x + threadIdx.x;
  if (idx >= NV * C) return;
  int c = idx % C, vi = idx / C;
  int p0[3]; float w[3];
  for (int d = 0; d < 3; ++d) {
    float p = (verts[vi * 3 + d] + 1.0f) * 0.5f * (float)(S - 1);
    p = fminf(fmaxf(p, 0.0f), (float)(S - 1));
    int q = (int)floorf(p);
    if (q > S - 2) q = S - 2;
    if (q < 0) q = 0;
    p0[d] = q; w[d] = p - (float)q;
  }
  float acc = 0.0f;
  for (int dz = 0; dz < 2; ++dz)
    for (int dy = 0; dy < 2; ++dy)
      for (int dx = 0; dx < 2; ++dx) {
        int iz = p0[0] + dz, iy = p0[1] + dy, ix = p0[2] + dx;
        float wt = (dz ? w[0] : 1.0f - w[0]) * (dy ? w[1] : 1.0f - w[1]) *
                   (dx ? w[2] : 1.0f - w[2]);
        acc += wt * (float)feat[(size_t)((iz * S + iy) * S + ix) * CS + c];
      }
  out[(size_t)vi * outStride + outOfs + c] = acc;
}

// ---------------------------------------------------------------------------
// Graph adjacency: dense A (N x N), A = clip(scatter + scatter^T,0,1) + I
// ---------------------------------------------------------------------------
__global__ void zero_f32_k(float* p, int n) {
  int i = blockIdx.x * blockDim.x + threadIdx.x;
  if (i < n) p[i] = 0.0f;
}
__global__ void adj_scatter_k(const int* __restrict__ faces, int nf, float* A, int N) {
  int idx = blockIdx.x * blockDim.x + threadIdx.x;
  if (idx >= nf * 3) return;
  int f = idx / 3, e = idx % 3;
  int i = faces[f * 3 + e];
  int j = faces[f * 3 + (e + 1) % 3];
  A[(size_t)i * N + j] = 1.0f;   // benign write races: all write 1.0f
  A[(size_t)j * N + i] = 1.0f;
}
__global__ void adj_finalize_k(float* A, float* deg, int N) {
  int i = blockIdx.x * blockDim.x + threadIdx.x;
  if (i >= N) return;
  A[(size_t)i * N + i] = 1.0f;
  float s = 0.0f;
  for (int j = 0; j < N; ++j) s += A[(size_t)i * N + j];
  deg[i] = s;
}

// t[i][f] = (A @ xf)[i][f] / deg[i]
__global__ void spmm_k(const float* __restrict__ A, const float* __restrict__ deg,
                       const float* __restrict__ xf, float* __restrict__ t, int N, int F) {
  int idx = blockIdx.x * blockDim.x + threadIdx.x;
  if (idx >= N * F) return;
  int f = idx % F, i = idx / F;
  float s = 0.0f;
  for (int j = 0; j < N; ++j) s += A[(size_t)i * N + j] * xf[(size_t)j * F + f];
  t[idx] = s / deg[i];
}

// y = xf@W0 + t@W1 + b  (optional relu)
__global__ void gemm_bias_k(const float* __restrict__ xf, const float* __restrict__ t,
                            const float* __restrict__ W0, const float* __restrict__ W1,
                            const float* __restrict__ bias, float* __restrict__ y,
                            int N, int Fin, int Fout, int relu) {
  int idx = blockIdx.x * blockDim.x + threadIdx.x;
  if (idx >= N * Fout) return;
  int o = idx % Fout, i = idx / Fout;
  float acc = bias[o];
  for (int f = 0; f < Fin; ++f) {
    acc += xf[(size_t)i * Fin + f] * W0[(size_t)f * Fout + o];
    acc += t[(size_t)i * Fin + f] * W1[(size_t)f * Fout + o];
  }
  if (relu) acc = acc > 0.0f ? acc : 0.0f;
  y[idx] = acc;
}

// ---------------------------------------------------------------------------
// small utility kernels
// ---------------------------------------------------------------------------
__global__ void copy_f32_k(const float* __restrict__ s, float* __restrict__ d, int n) {
  int i = blockIdx.x * blockDim.x + threadIdx.x;
  if (i < n) d[i] = s[i];
}
__global__ void copy_rows_k(const float* __restrict__ src, int ss, int so,
                            float* __restrict__ dst, int ds, int dofs, int rows, int cols) {
  int idx = blockIdx.x * blockDim.x + threadIdx.x;
  if (idx >= rows * cols) return;
  int c = idx % cols, r = idx / cols;
  dst[(size_t)r * ds + dofs + c] = src[(size_t)r * ss + so + c];
}
__global__ void add_k(float* __restrict__ d, const float* __restrict__ s, int n) {
  int i = blockIdx.x * blockDim.x + threadIdx.x;
  if (i < n) d[i] += s[i];
}
// rows[nprev+e] = 0.5*(rows[edge.a] + rows[edge.b]); reads rows < nprev only
__global__ void mid_rows_k(float* __restrict__ buf, int stride,
                           const int* __restrict__ edges, int nprev, int ne, int cols) {
  int idx = blockIdx.x * blockDim.x + threadIdx.x;
  if (idx >= ne * cols) return;
  int c = idx % cols, e = idx / cols;
  int a = edges[e * 2], b = edges[e * 2 + 1];
  buf[(size_t)(nprev + e) * stride + c] =
      0.5f * (buf[(size_t)a * stride + c] + buf[(size_t)b * stride + c]);
}

// point-to-segment deformation distance per edge
__global__ void dist_k(const float* __restrict__ v, const int* __restrict__ edges,
                       float* __restrict__ out, int nprev, int ne) {
  int e = blockIdx.x * blockDim.x + threadIdx.x;
  if (e >= ne) return;
  int ia = edges[e * 2], ib = edges[e * 2 + 1];
  float ax = v[ia * 3], ay = v[ia * 3 + 1], az = v[ia * 3 + 2];
  float bx = v[ib * 3], by = v[ib * 3 + 1], bz = v[ib * 3 + 2];
  float sx = v[(nprev + e) * 3], sy = v[(nprev + e) * 3 + 1], sz = v[(nprev + e) * 3 + 2];
  float vax = sx - ax, vay = sy - ay, vaz = sz - az;
  float vbx = sx - bx, vby = sy - by, vbz = sz - bz;
  float bax = bx - ax, bay = by - ay, baz = bz - az;
  float c1 = vax * bax + vay * bay + vaz * baz;
  float c2 = vbx * bax + vby * bay + vbz * baz;
  float cx = vay * baz - vaz * bay;
  float cy = vaz * bax - vax * baz;
  float cz = vax * bay - vay * bax;
  float d = sqrtf(cx * cx + cy * cy + cz * cz) / sqrtf(bax * bax + bay * bay + baz * baz);
  if (c1 < 0.0f) d = sqrtf(vax * vax + vay * vay + vaz * vaz);
  if (c2 < 0.0f) d = sqrtf(vbx * vbx + vby * vby + vbz * vbz);
  out[e] = d;
}

// ===========================================================================
// host orchestration
// ===========================================================================
extern "C" void kernel_launch(void* const* d_in, const int* in_sizes, int n_in,
                              void* d_out, int out_size, void* d_ws, size_t ws_size,
                              hipStream_t stream) {
  (void)in_sizes; (void)n_in; (void)out_size; (void)ws_size;
  const int T = 256;
  const int N64 = 64 * 64 * 64, N32 = 32 * 32 * 32, N16 = 16 * 16 * 16;
  float* outf = (float*)d_out;              // [v(486) | voxel_pred(2*N64) | dist(120)]
  float* out_vox = outf + 486;
  float* out_dist = outf + 486 + 2 * N64;

  // -------- deterministic scratch layout --------
  char* ws = (char*)d_ws;
  size_t off = 0;
  auto alloc = [&](size_t bytes) -> char* {
    char* p = ws + off;
    off = (off + bytes + 255) & ~(size_t)255;
    return p;
  };
  _Float16* a0  = (_Float16*)alloc((size_t)N64 * 1 * 2);
  _Float16* e0a = (_Float16*)alloc((size_t)N64 * 16 * 2);
  _Float16* d0c = (_Float16*)alloc((size_t)N64 * 32 * 2);   // [x0 | up1(h1)]
  _Float16* p1  = (_Float16*)alloc((size_t)N32 * 16 * 2);
  _Float16* e1a = (_Float16*)alloc((size_t)N32 * 32 * 2);
  _Float16* d1c = (_Float16*)alloc((size_t)N32 * 64 * 2);   // [x1 | up0(x2)]
  _Float16* p2  = (_Float16*)alloc((size_t)N16 * 32 * 2);
  _Float16* e2a = (_Float16*)alloc((size_t)N16 * 64 * 2);
  _Float16* x2b = (_Float16*)alloc((size_t)N16 * 64 * 2);
  _Float16* u0a = (_Float16*)alloc((size_t)N32 * 32 * 2);
  _Float16* h1  = (_Float16*)alloc((size_t)N32 * 32 * 2);
  _Float16* u1a = (_Float16*)alloc((size_t)N64 * 16 * 2);
  _Float16* h0  = (_Float16*)alloc((size_t)N64 * 16 * 2);

  auto wbytes = [](int Cin, int Cout) -> size_t {
    return (size_t)27 * ((Cin + 31) / 32) * (Cout / 16) * 32 * 16 * 2;
  };
  _Float16* wp_e0a = (_Float16*)alloc(wbytes(1, 16));
  _Float16* wp_e0b = (_Float16*)alloc(wbytes(16, 16));
  _Float16* wp_e1a = (_Float16*)alloc(wbytes(16, 32));
  _Float16* wp_e1b = (_Float16*)alloc(wbytes(32, 32));
  _Float16* wp_e2a = (_Float16*)alloc(wbytes(32, 64));
  _Float16* wp_e2b = (_Float16*)alloc(wbytes(64, 64));
  _Float16* wp_u0a = (_Float16*)alloc(wbytes(64, 32));
  _Float16* wp_u0b = (_Float16*)alloc(wbytes(32, 32));
  _Float16* wp_u1a = (_Float16*)alloc(wbytes(32, 16));
  _Float16* wp_u1b = (_Float16*)alloc(wbytes(16, 16));

  float* A0   = (float*)alloc(42 * 42 * 4);
  float* deg0 = (float*)alloc(42 * 4);
  float* A1   = (float*)alloc(162 * 162 * 4);
  float* deg1 = (float*)alloc(162 * 4);
  float* vcur  = (float*)alloc(162 * 3 * 4);
  float* latIn = (float*)alloc(162 * 67 * 4);
  float* latP  = (float*)alloc(162 * 32 * 4);
  float* latQ  = (float*)alloc(162 * 32 * 4);
  float* tsp   = (float*)alloc(162 * 67 * 4);
  float* g16   = (float*)alloc(162 * 16 * 4);
  float* g8    = (float*)alloc(162 * 8 * 4);
  float* dvb   = (float*)alloc(162 * 3 * 4);

  const int* edges  = (const int*)d_in[IDX_EDGES];
  const int* faces0 = (const int*)d_in[IDX_FACES0];
  const int* faces1 = (const int*)d_in[IDX_FACES1];

  // -------- weight prepack + input conversion --------
  auto pack = [&](int widx, _Float16* dst, int Cin, int Cout) {
    int total = 27 * ((Cin + 31) / 32) * (Cout / 16) * 32;
    pack_w_k<<<cdiv(total, T), T, 0, stream>>>((const float*)d_in[widx], dst, Cin, Cout);
  };
  pack(IDX_E0_WA, wp_e0a, 1, 16);   pack(IDX_E0_WB, wp_e0b, 16, 16);
  pack(IDX_E1_WA, wp_e1a, 16, 32);  pack(IDX_E1_WB, wp_e1b, 32, 32);
  pack(IDX_E2_WA, wp_e2a, 32, 64);  pack(IDX_E2_WB, wp_e2b, 64, 64);
  pack(IDX_U0_WA, wp_u0a, 64, 32);  pack(IDX_U0_WB, wp_u0b, 32, 32);
  pack(IDX_U1_WA, wp_u1a, 32, 16);  pack(IDX_U1_WB, wp_u1b, 16, 16);
  f32_to_f16_k<<<cdiv(N64, T), T, 0, stream>>>((const float*)d_in[IDX_X], a0, N64);

#define CONV(S, CIN, COUT, INS, RELU, ACT, WP, BIDX, OUT, OUTS, OOFS)                  \
  conv3d_wmma_k<S, CIN, COUT, INS, RELU>                                               \
      <<<cdiv(((S * S * S) / 16) * ((COUT) / 16), 8), 256, 0, stream>>>(               \
          ACT, WP, (const float*)d_in[BIDX], OUT, OUTS, OOFS)

  // -------- encoder --------
  CONV(64, 1, 16, 1, true, a0, wp_e0a, IDX_E0_BA, e0a, 16, 0);
  CONV(64, 16, 16, 16, true, e0a, wp_e0b, IDX_E0_BB, d0c, 32, 0);        // x0 -> d0c[0:16]
  maxpool_k<<<cdiv(N32 * 16, T), T, 0, stream>>>(d0c, 32, 16, 64, p1);
  CONV(32, 16, 32, 16, true, p1, wp_e1a, IDX_E1_BA, e1a, 32, 0);
  CONV(32, 32, 32, 32, true, e1a, wp_e1b, IDX_E1_BB, d1c, 64, 0);        // x1 -> d1c[0:32]
  maxpool_k<<<cdiv(N16 * 32, T), T, 0, stream>>>(d1c, 64, 32, 32, p2);
  CONV(16, 32, 64, 32, true, p2, wp_e2a, IDX_E2_BA, e2a, 64, 0);
  CONV(16, 64, 64, 64, true, e2a, wp_e2b, IDX_E2_BB, x2b, 64, 0);        // x2

  // -------- decoder stage 0 (32^3) --------
  upconv_k<<<cdiv(N32 * 32, T), T, 0, stream>>>(x2b, 64, 16,
      (const float*)d_in[IDX_U0_WT], (const float*)d_in[IDX_U0_BT], 32, d1c, 64, 32);
  CONV(32, 64, 32, 64, true, d1c, wp_u0a, IDX_U0_BA, u0a, 32, 0);
  CONV(32, 32, 32, 32, true, u0a, wp_u0b, IDX_U0_BB, h1, 32, 0);         // h1

  // -------- decoder stage 1 (64^3) --------
  upconv_k<<<cdiv(N64 * 16, T), T, 0, stream>>>(h1, 32, 32,
      (const float*)d_in[IDX_U1_WT], (const float*)d_in[IDX_U1_BT], 16, d0c, 32, 16);
  CONV(64, 32, 16, 32, true, d0c, wp_u1a, IDX_U1_BA, u1a, 16, 0);
  CONV(64, 16, 16, 16, true, u1a, wp_u1b, IDX_U1_BB, h0, 16, 0);         // h0
  final_conv_k<<<cdiv(2 * N64, T), T, 0, stream>>>(h0, (const float*)d_in[IDX_FIN_W],
                                                   (const float*)d_in[IDX_FIN_B],
                                                   out_vox, N64);

  // -------- adjacency --------
  zero_f32_k<<<cdiv(42 * 42, T), T, 0, stream>>>(A0, 42 * 42);
  adj_scatter_k<<<cdiv(80 * 3, T), T, 0, stream>>>(faces0, 80, A0, 42);
  adj_finalize_k<<<1, 64, 0, stream>>>(A0, deg0, 42);
  zero_f32_k<<<cdiv(162 * 162, T), T, 0, stream>>>(A1, 162 * 162);
  adj_scatter_k<<<cdiv(320 * 3, T), T, 0, stream>>>(faces1, 320, A1, 162);
  adj_finalize_k<<<1, 192, 0, stream>>>(A1, deg1, 162);

  auto gconv = [&](const float* A, const float* deg, int N, const float* xin, int Fin,
                   int base, int Fout, float* yout, int relu) {
    spmm_k<<<cdiv(N * Fin, T), T, 0, stream>>>(A, deg, xin, tsp, N, Fin);
    gemm_bias_k<<<cdiv(N * Fout, T), T, 0, stream>>>(
        xin, tsp, (const float*)d_in[base], (const float*)d_in[base + 1],
        (const float*)d_in[base + 2], yout, N, Fin, Fout, relu);
  };
  auto F2F = [](int s, int l) { return IDX_F2F + (s * 3 + l) * 3; };
  auto F2V = [](int s, int l) { return IDX_F2V + (s * 3 + l) * 3; };

  // -------- GNN stage 0 (N=42, feats 67 = 64 skip + 3 v) --------
  copy_f32_k<<<1, 128, 0, stream>>>((const float*)d_in[IDX_VERTS0], vcur, 126);
  trilinear_k<<<cdiv(42 * 64, T), T, 0, stream>>>(x2b, 16, 64, 64, vcur, 42, latIn, 67, 0);
  copy_rows_k<<<1, 128, 0, stream>>>(vcur, 3, 0, latIn, 67, 64, 42, 3);
  gconv(A0, deg0, 42, latIn, 67, F2F(0, 0), 32, latP, 1);
  gconv(A0, deg0, 42, latP, 32, F2F(0, 1), 32, latQ, 1);
  gconv(A0, deg0, 42, latQ, 32, F2F(0, 2), 32, latP, 1);   // lat0 = latP
  gconv(A0, deg0, 42, latP, 32, F2V(0, 0), 16, g16, 1);
  gconv(A0, deg0, 42, g16, 16, F2V(0, 1), 8, g8, 1);
  gconv(A0, deg0, 42, g8, 8, F2V(0, 2), 3, dvb, 0);
  add_k<<<1, 128, 0, stream>>>(vcur, dvb, 126);

  // -------- subdivide: midpoints of verts and latents --------
  mid_rows_k<<<cdiv(120 * 3, T), T, 0, stream>>>(vcur, 3, edges, 42, 120, 3);
  mid_rows_k<<<cdiv(120 * 32, T), T, 0, stream>>>(latP, 32, edges, 42, 120, 32);

  // -------- GNN stage 1 (N=162, feats 67 = 32 lat + 32 skip + 3 v) --------
  copy_rows_k<<<cdiv(162 * 32, T), T, 0, stream>>>(latP, 32, 0, latIn, 67, 0, 162, 32);
  trilinear_k<<<cdiv(162 * 32, T), T, 0, stream>>>(h1, 32, 32, 32, vcur, 162, latIn, 67, 32);
  copy_rows_k<<<cdiv(162 * 3, T), T, 0, stream>>>(vcur, 3, 0, latIn, 67, 64, 162, 3);
  gconv(A1, deg1, 162, latIn, 67, F2F(1, 0), 32, latQ, 1);
  gconv(A1, deg1, 162, latQ, 32, F2F(1, 1), 32, latP, 1);
  gconv(A1, deg1, 162, latP, 32, F2F(1, 2), 32, latQ, 1);  // lat1 = latQ
  gconv(A1, deg1, 162, latQ, 32, F2V(1, 0), 16, g16, 1);
  gconv(A1, deg1, 162, g16, 16, F2V(1, 1), 8, g8, 1);
  gconv(A1, deg1, 162, g8, 8, F2V(1, 2), 3, dvb, 0);
  add_k<<<cdiv(486, T), T, 0, stream>>>(vcur, dvb, 486);
  dist_k<<<1, 128, 0, stream>>>(vcur, edges, out_dist, 42, 120);  // snapshot at v2

  // -------- GNN stage 2 (N=162, feats 51 = 32 lat + 16 skip + 3 v) --------
  copy_rows_k<<<cdiv(162 * 32, T), T, 0, stream>>>(latQ, 32, 0, latIn, 51, 0, 162, 32);
  trilinear_k<<<cdiv(162 * 16, T), T, 0, stream>>>(h0, 64, 16, 16, vcur, 162, latIn, 51, 32);
  copy_rows_k<<<cdiv(162 * 3, T), T, 0, stream>>>(vcur, 3, 0, latIn, 51, 48, 162, 3);
  gconv(A1, deg1, 162, latIn, 51, F2F(2, 0), 32, latP, 1);
  gconv(A1, deg1, 162, latP, 32, F2F(2, 1), 32, latQ, 1);
  gconv(A1, deg1, 162, latQ, 32, F2F(2, 2), 32, latP, 1);
  gconv(A1, deg1, 162, latP, 32, F2V(2, 0), 16, g16, 1);
  gconv(A1, deg1, 162, g16, 16, F2V(2, 1), 8, g8, 1);
  gconv(A1, deg1, 162, g8, 8, F2V(2, 2), 3, dvb, 0);
  add_k<<<cdiv(486, T), T, 0, stream>>>(vcur, dvb, 486);
  copy_f32_k<<<cdiv(486, T), T, 0, stream>>>(vcur, outf, 486);   // final vertices

#undef CONV
}
